// MeshAutoencoder_14388140442207
// MI455X (gfx1250) — matrix operand order repaired and checked by reading, not compiled, not run on hardware
//
#include <hip/hip_runtime.h>
#include <hip/hip_bf16.h>

typedef __attribute__((ext_vector_type(16))) _Float16 v16h;
typedef __attribute__((ext_vector_type(8)))  float    v8f;
typedef __attribute__((ext_vector_type(8)))  _Float16 h8;
typedef unsigned int u32x4 __attribute__((ext_vector_type(4)));
typedef int          i32x4 __attribute__((ext_vector_type(4)));
typedef int          i32x8 __attribute__((ext_vector_type(8)));
typedef _Float16 half_t;

#define B_     2
#define NV_    4096
#define NF_    8192
#define NE_    32768
#define DIM_   512
#define ND_    128
#define DCE_   64
#define DCB_   192
#define NCODE_ 16384
#define NROW_  (B_*NF_)   /* 16384 face rows   */
#define VROW_  (B_*NV_)   /* 8192 vertex rows  */

// ---------------------------------------------------------------- WMMA helpers

static __device__ __forceinline__ v16h mk_frag(h8 lo, h8 hi) {
  v16h f;
#pragma unroll
  for (int e = 0; e < 8; ++e) { f[e] = lo[e]; f[e + 8] = hi[e]; }
  return f;
}

static __device__ __forceinline__ v8f wmma_f16(v16h a, v16h b, v8f c) {
  return __builtin_amdgcn_wmma_f32_16x16x32_f16(false, a, false, b, (short)0, c,
                                                false, false);
}

// ---------------------------------------------------------------- TDM tile load
// 2D Tensor-DMA descriptor (ISA ch.8 D# groups 0/1): move a [rows x 32] f16
// tile with row stride K (elements) from global into LDS at ldsoff, contiguous.
static __device__ __forceinline__ void tdm_load_tile_a(const half_t* gsrc,
                                                       unsigned ldsoff,
                                                       int rows, int K) {
  unsigned long long ga = (unsigned long long)(size_t)gsrc;
  u32x4 g0;
  g0[0] = 1u;                                             // count=1, user D#
  g0[1] = ldsoff;                                         // lds_addr (bytes)
  g0[2] = (unsigned)ga;                                   // global_addr lo
  g0[3] = (unsigned)((ga >> 32) & 0x01ffffffu) | (2u << 30);  // addr hi | type=2
  i32x8 g1;
  g1[0] = (int)(1u << 16);                                // data_size = 2 bytes
  g1[1] = (int)(((unsigned)K & 0xffffu) << 16);           // tensor_dim0[15:0]
  g1[2] = (int)((((unsigned)K >> 16) & 0xffffu) |
                (((unsigned)rows & 0xffffu) << 16));      // dim0 hi | dim1 lo
  g1[3] = (int)(32u << 16);                               // dim1 hi=0 | tile_dim0=32
  g1[4] = (int)(rows & 0xffff);                           // tile_dim1 | tile_dim2=0
  g1[5] = (int)K;                                         // tensor_dim0_stride lo
  g1[6] = 0;                                              // stride hi | dim1_stride lo
  g1[7] = 0;                                              // dim1_stride hi
  i32x4 z4 = {0, 0, 0, 0};                                // 2D tensor: groups 2/3 unused
  i32x8 z8 = {0, 0, 0, 0, 0, 0, 0, 0};
  __builtin_amdgcn_tensor_load_to_lds(g0, g1, z4, z4, z8, 0);
}

// ---------------------------------------------------------------- GEMM (f16 in, f32 out)
// C[M,N] (+)= A[M,K] @ B[K,N] (+ bias).  M%128==0, N%64==0, K%32==0.
// A tile staged by the Tensor Data Mover; B tile transposed on store.
__global__ __launch_bounds__(256) void k_gemm_f16(
    const half_t* __restrict__ A, const half_t* __restrict__ Bm,
    float* __restrict__ C, half_t* __restrict__ Ch,
    const float* __restrict__ bias, int M, int N, int K, int accum)
{
  __shared__ half_t As[128 * 32];   // [row][k]  (TDM destination)
  __shared__ half_t Bs[64 * 32];    // transposed: [n][k]

  const int tid  = threadIdx.x;
  const int lane = tid & 31;
  const int wid  = tid >> 5;
  const int wm   = wid & 3;     // 4 waves over M (32 rows each)
  const int wn   = wid >> 2;    // 2 waves over N (32 cols each)
  const int rlo  = lane & 15;
  const int kh   = lane >> 4;

  const int ntn = N >> 6;
  const int m0  = (blockIdx.x / ntn) * 128;
  const int n0  = (blockIdx.x % ntn) * 64;

  const unsigned ldsA = (unsigned)(size_t)(void*)As;  // LDS byte offset

  v8f zero = {};
  v8f acc[2][2];
  acc[0][0] = zero; acc[0][1] = zero; acc[1][0] = zero; acc[1][1] = zero;

  const int bk = tid >> 3, bng = tid & 7;  // B tile: 32 k x 64 n

  for (int k0 = 0; k0 < K; k0 += 32) {
    if (wid == 0)                            // one TDM issue per workgroup
      tdm_load_tile_a(A + (size_t)m0 * K + k0, ldsA, 128, K);
    {
      const h8 v = *(const h8*)(Bm + (size_t)(k0 + bk) * N + n0 + bng * 8);
#pragma unroll
      for (int e = 0; e < 8; ++e) Bs[(bng * 8 + e) * 32 + bk] = v[e];
    }
    if (k0 + 32 < K && lane == 0)            // prefetch next B tile rows
      __builtin_prefetch(Bm + (size_t)(k0 + 32 + wid * 4) * N + n0, 0, 1);
    if (wid == 0)
      __builtin_amdgcn_s_wait_tensorcnt(0);  // TDM done before barrier signal
    __syncthreads();

    v16h af[2], bf[2];
#pragma unroll
    for (int mi = 0; mi < 2; ++mi) {   // A frag: interleaved K pattern
      const h8* pa = (const h8*)(As + (wm * 32 + mi * 16 + rlo) * 32);
      af[mi] = mk_frag(pa[kh], pa[2 + kh]);
    }
#pragma unroll
    for (int ni = 0; ni < 2; ++ni) {   // B frag: interleaved K pattern
      const h8* pb = (const h8*)(Bs + (wn * 32 + ni * 16 + rlo) * 32);
      bf[ni] = mk_frag(pb[kh], pb[2 + kh]);
    }
#pragma unroll
    for (int mi = 0; mi < 2; ++mi)
#pragma unroll
      for (int ni = 0; ni < 2; ++ni)
        acc[mi][ni] = wmma_f16(af[mi], bf[ni], acc[mi][ni]);
    __syncthreads();
  }

#pragma unroll
  for (int mi = 0; mi < 2; ++mi) {
#pragma unroll
    for (int ni = 0; ni < 2; ++ni) {
      const int col = n0 + wn * 32 + ni * 16 + rlo;
      const float bv = bias ? bias[col] : 0.f;
#pragma unroll
      for (int v = 0; v < 8; ++v) {
        const int row = m0 + wm * 32 + mi * 16 + v + 8 * kh;
        const size_t off = (size_t)row * N + col;
        float val = acc[mi][ni][v] + bv;
        if (accum) val += C[off];
        C[off] = val;
        if (Ch) Ch[off] = (half_t)val;
      }
    }
  }
}

// ---------------------------------------------------------------- fused VQ argmin
// For 16 residual rows per block, sweep the whole codebook with WMMA and track
// argmin of (|c|^2 - 2 r.c)   (|r|^2 is row-constant -> irrelevant for argmin).
__global__ __launch_bounds__(128) void k_vq_argmin(
    const half_t* __restrict__ R,      // [VROW, 192] f16 residual
    const half_t* __restrict__ CB,     // [NCODE, 192] f16 codebook
    const float*  __restrict__ cbsq,   // [NCODE]
    int* __restrict__ ind)
{
  const int tid  = threadIdx.x;
  const int lane = tid & 31;
  const int wid  = tid >> 5;   // 4 waves, interleaved over codebook chunks
  const int rlo  = lane & 15;
  const int kh   = lane >> 4;
  const int m0   = blockIdx.x * 16;

  v16h afr[6];                 // K=192 -> 6 fragments, kept in registers
  {
    const h8* pr = (const h8*)(R + (size_t)(m0 + rlo) * DCB_);
#pragma unroll
    for (int t = 0; t < 6; ++t)
      afr[t] = mk_frag(pr[t * 4 + kh], pr[t * 4 + 2 + kh]);
  }

  float minv[8]; int mini[8];
#pragma unroll
  for (int v = 0; v < 8; ++v) { minv[v] = 3.0e38f; mini[v] = 0; }

  for (int n0 = wid * 16; n0 < NCODE_; n0 += 64) {
    const int crow = n0 + rlo;                    // this lane's codebook row
    const h8* pc = (const h8*)(CB + (size_t)crow * DCB_);
    v8f acc = {};
#pragma unroll
    for (int t = 0; t < 6; ++t) {
      v16h bfr = mk_frag(pc[t * 4 + kh], pc[t * 4 + 2 + kh]);
      acc = wmma_f16(afr[t], bfr, acc);
    }
    const float cq = cbsq[crow];
#pragma unroll
    for (int v = 0; v < 8; ++v) {
      const float s = cq - 2.0f * acc[v];
      if (s < minv[v]) { minv[v] = s; mini[v] = crow; }
    }
  }

  __shared__ float sv[16 * 64];
  __shared__ int   si[16 * 64];
#pragma unroll
  for (int v = 0; v < 8; ++v) {
    const int row  = v + 8 * kh;
    const int cand = wid * 16 + rlo;
    sv[row * 64 + cand] = minv[v];
    si[row * 64 + cand] = mini[v];
  }
  __syncthreads();
  if (tid < 16) {
    float bv = 3.0e38f; int bi = 0;
    for (int c = 0; c < 64; ++c) {
      float v = sv[tid * 64 + c];
      if (v < bv) { bv = v; bi = si[tid * 64 + c]; }
    }
    ind[m0 + tid] = bi;
  }
}

// ---------------------------------------------------------------- elementwise / glue

__global__ void k_cast16(const float* __restrict__ x, half_t* __restrict__ y, size_t n) {
  size_t i = (size_t)blockIdx.x * blockDim.x + threadIdx.x;
  if (i < n) y[i] = (half_t)x[i];
}

// conv_w [2,2,O=512,I=512,K=3] -> W2h [layer][k*512+ci][co]  (im2col weights, f16)
__global__ void k_wpack(const float* __restrict__ w, half_t* __restrict__ o) {
  size_t gid = (size_t)blockIdx.x * blockDim.x + threadIdx.x;
  if (gid >= (size_t)4 * 1536 * 512) return;
  int layer = (int)(gid / (1536 * 512));
  int rem   = (int)(gid % (1536 * 512));
  int kc = rem / 512, co = rem % 512;
  int k = kc / 512, ci = kc % 512;
  o[gid] = (half_t)w[(((size_t)layer * 512 + co) * 512 + ci) * 3 + k];
}

__global__ void k_cbsq(const float* __restrict__ cb, float* __restrict__ sq) {
  int i = blockIdx.x * blockDim.x + threadIdx.x;
  if (i >= NCODE_) return;
  const float* r = cb + (size_t)i * DCB_;
  float s = 0.f;
  for (int d = 0; d < DCB_; ++d) s += r[d] * r[d];
  sq[i] = s;
}

__global__ void k_disc(const float* __restrict__ v, int* __restrict__ d, int n) {
  int i = blockIdx.x * blockDim.x + threadIdx.x;
  if (i >= n) return;
  float t = (v[i] + 1.0f) * 0.5f * (float)ND_ - 0.5f;
  float r = rintf(t);
  r = fminf(fmaxf(r, 0.f), (float)(ND_ - 1));
  d[i] = (int)r;
}

__global__ void k_face_embed(const int* __restrict__ faces, const int* __restrict__ disc,
                             const float* __restrict__ emb, int* __restrict__ fc,
                             half_t* __restrict__ feh) {
  int gid = blockIdx.x * blockDim.x + threadIdx.x;
  if (gid >= NROW_ * 9) return;
  int row = gid / 9, s = gid % 9;
  int b = row / NF_;
  int j = s / 3, t = s % 3;
  int vid = faces[row * 3 + j];
  int c = disc[(b * NV_ + vid) * 3 + t];
  fc[gid] = c;
  const float* e = emb + c * DCE_;
  half_t* o = feh + (size_t)row * (9 * DCE_) + s * DCE_;
#pragma unroll 8
  for (int d = 0; d < DCE_; ++d) o[d] = (half_t)e[d];
}

__global__ void k_edge_cnt(const int* __restrict__ ed, float* __restrict__ cnt) {
  int i = blockIdx.x * blockDim.x + threadIdx.x;
  if (i >= B_ * NE_) return;
  int b = i / NE_, e = i % NE_;
  int dst = ed[(b * 2 + 1) * NE_ + e];
  atomicAdd(&cnt[b * NF_ + dst], 1.0f);
}

__global__ void k_edge_aggr(const int* __restrict__ ed, const float* __restrict__ x,
                            float* __restrict__ aggr) {
  long gid = (long)blockIdx.x * blockDim.x + threadIdx.x;
  if (gid >= (long)B_ * NE_ * DIM_) return;
  int d = (int)(gid % DIM_);
  long ei = gid / DIM_;
  int b = (int)(ei / NE_), e = (int)(ei % NE_);
  int src = ed[(b * 2 + 0) * NE_ + e];
  int dst = ed[(b * 2 + 1) * NE_ + e];
  atomicAdd(&aggr[(size_t)(b * NF_ + dst) * DIM_ + d],
            x[(size_t)(b * NF_ + src) * DIM_ + d]);
}

__global__ void k_aggr_norm(const float* __restrict__ aggr, const float* __restrict__ cnt,
                            half_t* __restrict__ out) {
  size_t gid = (size_t)blockIdx.x * blockDim.x + threadIdx.x;
  if (gid >= (size_t)NROW_ * DIM_) return;
  float c = fmaxf(cnt[gid / DIM_], 1.0f);
  out[gid] = (half_t)(aggr[gid] / c);
}

__global__ void k_scatter_cnt(const int* __restrict__ faces, float* __restrict__ vcnt) {
  int i = blockIdx.x * blockDim.x + threadIdx.x;
  if (i >= B_ * NF_ * 3) return;
  int b = i / (NF_ * 3);
  int vid = faces[i];
  atomicAdd(&vcnt[b * NV_ + vid], 1.0f);
}

__global__ void k_scatter_add(const int* __restrict__ faces, const float* __restrict__ fe2,
                              float* __restrict__ vsum) {
  long gid = (long)blockIdx.x * blockDim.x + threadIdx.x;
  if (gid >= (long)B_ * NF_ * 3 * DCB_) return;
  int d = (int)(gid % DCB_);
  long slot = gid / DCB_;            // (b*NF + f)*3 + j
  int b = (int)(slot / (NF_ * 3));
  long rem = slot % (NF_ * 3);
  int f = (int)(rem / 3), j = (int)(rem % 3);
  int vid = faces[slot];
  atomicAdd(&vsum[((size_t)(b * NV_ + vid)) * DCB_ + d],
            fe2[((size_t)(b * NF_ + f)) * (3 * DCB_) + j * DCB_ + d]);
}

__global__ void k_vq_init(const float* __restrict__ vsum, const float* __restrict__ vcnt,
                          float* __restrict__ res, half_t* __restrict__ resh,
                          float* __restrict__ quant) {
  size_t gid = (size_t)blockIdx.x * blockDim.x + threadIdx.x;
  if (gid >= (size_t)VROW_ * DCB_) return;
  float c = fmaxf(vcnt[gid / DCB_], 1e-5f);
  float r = vsum[gid] / c;
  res[gid] = r;
  resh[gid] = (half_t)r;
  quant[gid] = 0.f;
}

__global__ __launch_bounds__(256) void k_vq_update(
    const float* __restrict__ cb, const int* __restrict__ ind,
    float* __restrict__ residual, half_t* __restrict__ residualh,
    float* __restrict__ quant, float* __restrict__ commit_acc) {
  __shared__ float red[256];
  size_t gid = (size_t)blockIdx.x * blockDim.x + threadIdx.x;
  float dd = 0.f;
  if (gid < (size_t)VROW_ * DCB_) {
    int row = (int)(gid / DCB_);
    int d = (int)(gid % DCB_);
    float qv = cb[(size_t)ind[row] * DCB_ + d];
    float r = residual[gid];
    float diff = qv - r;
    dd = diff * diff;
    quant[gid] += qv;
    float nr = r - qv;
    residual[gid] = nr;
    residualh[gid] = (half_t)nr;
  }
  int tid = threadIdx.x;
  red[tid] = dd;
  __syncthreads();
  for (int s = 128; s > 0; s >>= 1) {
    if (tid < s) red[tid] += red[tid + s];
    __syncthreads();
  }
  if (tid == 0) atomicAdd(commit_acc, red[0]);
}

__global__ void k_gather_quant(const int* __restrict__ faces, const float* __restrict__ quant,
                               half_t* __restrict__ fqh) {
  long gid = (long)blockIdx.x * blockDim.x + threadIdx.x;
  if (gid >= (long)B_ * NF_ * 3 * DCB_) return;
  int d = (int)(gid % DCB_);
  long slot = gid / DCB_;
  int b = (int)(slot / (NF_ * 3));
  long rem = slot % (NF_ * 3);
  int f = (int)(rem / 3), j = (int)(rem % 3);
  int vid = faces[slot];
  fqh[((size_t)(b * NF_ + f)) * (3 * DCB_) + j * DCB_ + d] =
      (half_t)quant[((size_t)(b * NV_ + vid)) * DCB_ + d];
}

__global__ void k_im2col(const float* __restrict__ y, half_t* __restrict__ col) {
  long gid = (long)blockIdx.x * blockDim.x + threadIdx.x;
  if (gid >= (long)NROW_ * 3 * DIM_) return;
  int c = (int)(gid % (3 * DIM_));
  int row = (int)(gid / (3 * DIM_));
  int k = c / DIM_, ci = c % DIM_;
  int b = row / NF_, f = row % NF_;
  int fs = f + k - 1;                       // SAME pad within each batch
  float v = (fs >= 0 && fs < NF_) ? y[((size_t)(b * NF_ + fs)) * DIM_ + ci] : 0.f;
  col[gid] = (half_t)v;
}

__global__ __launch_bounds__(256) void k_gn_stats(const float* __restrict__ y,
                                                  float* __restrict__ stats) {
  int bid = blockIdx.x;  // b*8 + g
  int b = bid >> 3, g = bid & 7;
  int tid = threadIdx.x;
  float s = 0.f, ss = 0.f;
  const int total = NF_ * 64;
  for (int i = tid; i < total; i += 256) {
    int f = i >> 6;
    int c = (g << 6) + (i & 63);
    float v = y[((size_t)(b * NF_ + f)) * DIM_ + c];
    s += v; ss += v * v;
  }
  __shared__ float rs[256], rss[256];
  rs[tid] = s; rss[tid] = ss;
  __syncthreads();
  for (int st = 128; st > 0; st >>= 1) {
    if (tid < st) { rs[tid] += rs[tid + st]; rss[tid] += rss[tid + st]; }
    __syncthreads();
  }
  if (tid == 0) {
    float mu = rs[0] / (float)total;
    float var = rss[0] / (float)total - mu * mu;
    stats[bid * 2] = mu;
    stats[bid * 2 + 1] = rsqrtf(var + 1e-5f);
  }
}

__global__ void k_gn_silu(const float* __restrict__ yin_, float* __restrict__ yout,
                          const float* __restrict__ stats, const float* __restrict__ w,
                          const float* __restrict__ bb) {
  size_t gid = (size_t)blockIdx.x * blockDim.x + threadIdx.x;
  if (gid >= (size_t)NROW_ * DIM_) return;
  int c = (int)(gid % DIM_);
  int row = (int)(gid / DIM_);
  int b = row / NF_;
  int g = c >> 6;
  float mu = stats[(b * 8 + g) * 2], rstd = stats[(b * 8 + g) * 2 + 1];
  float v = (yin_[gid] - mu) * rstd * w[c] + bb[c];
  yout[gid] = v / (1.f + expf(-v));
}

__global__ void k_add(float* __restrict__ a, const float* __restrict__ b, size_t n) {
  size_t i = (size_t)blockIdx.x * blockDim.x + threadIdx.x;
  if (i < n) a[i] += b[i];
}

// log-softmax over 128 bins + 3-tap gaussian-blurred one-hot target
__global__ __launch_bounds__(256) void k_ce(const float* __restrict__ logits,
                                            const int* __restrict__ fc,
                                            float* __restrict__ acc) {
  __shared__ float part[8];
  int tid = threadIdx.x, lane = tid & 31, wid = tid >> 5;
  long rr = (long)blockIdx.x * 8 + wid;      // row over B*NF*9
  float loss = 0.f;
  if (rr < (long)NROW_ * 9) {
    long base = (rr / 9) * (9 * ND_) + (rr % 9) * ND_;
    float x0 = logits[base + lane * 4 + 0];
    float x1 = logits[base + lane * 4 + 1];
    float x2 = logits[base + lane * 4 + 2];
    float x3 = logits[base + lane * 4 + 3];
    float m = fmaxf(fmaxf(x0, x1), fmaxf(x2, x3));
    for (int off = 16; off; off >>= 1) m = fmaxf(m, __shfl_xor(m, off));
    float s = expf(x0 - m) + expf(x1 - m) + expf(x2 - m) + expf(x3 - m);
    for (int off = 16; off; off >>= 1) s += __shfl_xor(s, off);
    if (lane == 0) {
      float lse = m + logf(s);
      int c = fc[rr];
      float e = expf(-3.125f);               // exp(-1/(2*0.4^2))
      float inv = 1.f / (1.f + 2.f * e);
      float g0 = e * inv, g1 = inv;
      float l = 0.f;
      if (c - 1 >= 0)   l += g0 * (lse - logits[base + c - 1]);
      l += g1 * (lse - logits[base + c]);
      if (c + 1 < ND_)  l += g0 * (lse - logits[base + c + 1]);
      loss = l;
    }
  }
  if (lane == 0) part[wid] = loss;
  __syncthreads();
  if (tid == 0) {
    float t = 0.f;
    for (int i = 0; i < 8; ++i) t += part[i];
    atomicAdd(acc, t);
  }
}

__global__ void k_final(const float* __restrict__ sc, float* __restrict__ out) {
  // sc[0], sc[1] = commit sums; sc[2] = recon sum
  float recon = sc[2] / (float)((long)NROW_ * 9);
  float commit = (sc[0] + sc[1]) / (float)((long)VROW_ * DCB_);
  out[0] = recon + 0.1f * commit;
}

// ---------------------------------------------------------------- launch

extern "C" void kernel_launch(void* const* d_in, const int* in_sizes, int n_in,
                              void* d_out, int out_size, void* d_ws, size_t ws_size,
                              hipStream_t stream) {
  (void)in_sizes; (void)n_in; (void)out_size; (void)ws_size;

  const float* vertices   = (const float*)d_in[0];
  const int*   faces      = (const int*)  d_in[1];
  const int*   face_edges = (const int*)  d_in[2];
  const float* coor_embed = (const float*)d_in[3];
  const float* W_in       = (const float*)d_in[4];
  const float* b_in       = (const float*)d_in[5];
  const float* sage_Wl    = (const float*)d_in[6];
  const float* sage_bl    = (const float*)d_in[7];
  const float* sage_Wr    = (const float*)d_in[8];
  const float* W_cb       = (const float*)d_in[9];
  const float* b_cb       = (const float*)d_in[10];
  const float* codebook   = (const float*)d_in[11];
  const float* W_out      = (const float*)d_in[12];
  const float* b_out      = (const float*)d_in[13];
  const float* conv_w     = (const float*)d_in[14];
  const float* conv_b     = (const float*)d_in[15];
  const float* gn_w       = (const float*)d_in[16];
  const float* gn_b       = (const float*)d_in[17];
  const float* W_logits   = (const float*)d_in[18];
  const float* b_logits   = (const float*)d_in[19];

  char* base = (char*)d_ws;
  size_t off = 0;
  auto alloc = [&](size_t bytes) -> void* {
    void* p = base + off;
    off += (bytes + 255) & ~(size_t)255;
    return p;
  };

  // f16 weight copies
  half_t* Winh  = (half_t*)alloc((size_t)576 * 512 * 2);
  half_t* Wlh   = (half_t*)alloc((size_t)2 * 512 * 512 * 2);
  half_t* Wrh   = (half_t*)alloc((size_t)2 * 512 * 512 * 2);
  half_t* Wcbh  = (half_t*)alloc((size_t)512 * 576 * 2);
  half_t* cbh   = (half_t*)alloc((size_t)NCODE_ * DCB_ * 2);
  half_t* Wouth = (half_t*)alloc((size_t)576 * 512 * 2);
  half_t* Wlogh = (half_t*)alloc((size_t)512 * 1152 * 2);
  half_t* W2h   = (half_t*)alloc((size_t)4 * 1536 * 512 * 2);
  float*  cbsq  = (float*)alloc((size_t)NCODE_ * 4);
  // activations
  int*    disc  = (int*)alloc((size_t)B_ * NV_ * 3 * 4);
  int*    fco   = (int*)alloc((size_t)NROW_ * 9 * 4);
  half_t* feh   = (half_t*)alloc((size_t)NROW_ * 576 * 2);   // aliased by fqh later
  float*  x     = (float*)alloc((size_t)NROW_ * DIM_ * 4);   // |
  float*  x2    = (float*)alloc((size_t)NROW_ * DIM_ * 4);   // | aliased by logits
  half_t* xh    = (half_t*)alloc((size_t)NROW_ * DIM_ * 2);  // |
  half_t* xh2   = (half_t*)alloc((size_t)NROW_ * DIM_ * 2);
  float*  aggr  = (float*)alloc((size_t)NROW_ * DIM_ * 4);   // | aliased by col
  half_t* aggrh = (half_t*)alloc((size_t)NROW_ * DIM_ * 2);  // |
  float*  cnt   = (float*)alloc((size_t)NROW_ * 4);
  float*  fe2   = (float*)alloc((size_t)NROW_ * 576 * 4);
  float*  vsum  = (float*)alloc((size_t)VROW_ * DCB_ * 4);
  float*  vcnt  = (float*)alloc((size_t)VROW_ * 4);
  float*  resid = (float*)alloc((size_t)VROW_ * DCB_ * 4);
  half_t* residh= (half_t*)alloc((size_t)VROW_ * DCB_ * 2);
  float*  quant = (float*)alloc((size_t)VROW_ * DCB_ * 4);
  int*    ind   = (int*)alloc((size_t)VROW_ * 4);
  float*  ydec  = (float*)alloc((size_t)NROW_ * DIM_ * 4);
  float*  yin   = (float*)alloc((size_t)NROW_ * DIM_ * 4);
  float*  ytmp  = (float*)alloc((size_t)NROW_ * DIM_ * 4);
  half_t* dech  = (half_t*)alloc((size_t)NROW_ * DIM_ * 2);
  float*  stats = (float*)alloc((size_t)B_ * 8 * 2 * 4);
  float*  scal  = (float*)alloc(4 * 4);
  // aliases (lifetimes disjoint)
  half_t* fqh    = feh;            // 16384x576 f16, same size as feh
  float*  logits = x;              // 16384x1152 f32 fits in x+x2+xh region
  half_t* col    = (half_t*)aggr;  // 16384x1536 f16 fits in aggr+aggrh region

  const int TPB = 256;
  auto blocks = [](size_t n) { return (unsigned)((n + 255) / 256); };
  auto gemm = [&](const half_t* A, const half_t* Bm, float* C, half_t* Ch,
                  const float* bias_, int M, int N, int K, int accum_) {
    k_gemm_f16<<<(M / 128) * (N / 64), 256, 0, stream>>>(A, Bm, C, Ch, bias_,
                                                         M, N, K, accum_);
  };

  // ---- weight preprocessing (deterministic each call) ----
  k_cast16<<<blocks(576 * 512), TPB, 0, stream>>>(W_in, Winh, (size_t)576 * 512);
  k_cast16<<<blocks(2 * 512 * 512), TPB, 0, stream>>>(sage_Wl, Wlh, (size_t)2 * 512 * 512);
  k_cast16<<<blocks(2 * 512 * 512), TPB, 0, stream>>>(sage_Wr, Wrh, (size_t)2 * 512 * 512);
  k_cast16<<<blocks(512 * 576), TPB, 0, stream>>>(W_cb, Wcbh, (size_t)512 * 576);
  k_cast16<<<blocks((size_t)NCODE_ * DCB_), TPB, 0, stream>>>(codebook, cbh, (size_t)NCODE_ * DCB_);
  k_cast16<<<blocks(576 * 512), TPB, 0, stream>>>(W_out, Wouth, (size_t)576 * 512);
  k_cast16<<<blocks(512 * 1152), TPB, 0, stream>>>(W_logits, Wlogh, (size_t)512 * 1152);
  k_wpack<<<blocks((size_t)4 * 1536 * 512), TPB, 0, stream>>>(conv_w, W2h);
  k_cbsq<<<blocks(NCODE_), TPB, 0, stream>>>(codebook, cbsq);

  // ---- discretize + face embedding + W_in projection ----
  k_disc<<<blocks(B_ * NV_ * 3), TPB, 0, stream>>>(vertices, disc, B_ * NV_ * 3);
  k_face_embed<<<blocks((size_t)NROW_ * 9), TPB, 0, stream>>>(faces, disc, coor_embed, fco, feh);
  gemm(feh, Winh, x, xh, b_in, NROW_, DIM_, 9 * DCE_, 0);

  // ---- SAGE message passing x2 ----
  (void)hipMemsetAsync(cnt, 0, (size_t)NROW_ * 4, stream);
  k_edge_cnt<<<blocks(B_ * NE_), TPB, 0, stream>>>(face_edges, cnt);
  float* xa = x; float* xb = x2; half_t* xha = xh; half_t* xhb = xh2;
  for (int i = 0; i < 2; ++i) {
    (void)hipMemsetAsync(aggr, 0, (size_t)NROW_ * DIM_ * 4, stream);
    k_edge_aggr<<<blocks((size_t)B_ * NE_ * DIM_), TPB, 0, stream>>>(face_edges, xa, aggr);
    k_aggr_norm<<<blocks((size_t)NROW_ * DIM_), TPB, 0, stream>>>(aggr, cnt, aggrh);
    gemm(aggrh, Wlh + (size_t)i * 512 * 512, xb, nullptr, sage_bl + i * 512,
         NROW_, DIM_, DIM_, 0);
    gemm(xha, Wrh + (size_t)i * 512 * 512, xb, xhb, nullptr, NROW_, DIM_, DIM_, 1);
    float* tf = xa; xa = xb; xb = tf;
    half_t* th = xha; xha = xhb; xhb = th;
  }

  // ---- project to codebook dim, scatter-mean onto vertices ----
  gemm(xha, Wcbh, fe2, nullptr, b_cb, NROW_, 3 * DCB_, DIM_, 0);
  (void)hipMemsetAsync(vsum, 0, (size_t)VROW_ * DCB_ * 4, stream);
  (void)hipMemsetAsync(vcnt, 0, (size_t)VROW_ * 4, stream);
  k_scatter_cnt<<<blocks(B_ * NF_ * 3), TPB, 0, stream>>>(faces, vcnt);
  k_scatter_add<<<blocks((size_t)B_ * NF_ * 3 * DCB_), TPB, 0, stream>>>(faces, fe2, vsum);
  k_vq_init<<<blocks((size_t)VROW_ * DCB_), TPB, 0, stream>>>(vsum, vcnt, resid, residh, quant);

  // ---- residual VQ x2 ----
  (void)hipMemsetAsync(scal, 0, 4 * 4, stream);
  for (int q = 0; q < 2; ++q) {
    k_vq_argmin<<<VROW_ / 16, 128, 0, stream>>>(residh, cbh, cbsq, ind);
    k_vq_update<<<blocks((size_t)VROW_ * DCB_), TPB, 0, stream>>>(
        codebook, ind, resid, residh, quant, scal + q);
  }

  // ---- gather back to faces, project out ----
  k_gather_quant<<<blocks((size_t)B_ * NF_ * 3 * DCB_), TPB, 0, stream>>>(faces, quant, fqh);
  gemm(fqh, Wouth, ydec, nullptr, b_out, NROW_, DIM_, 3 * DCB_, 0);

  // ---- decoder resnet blocks ----
  for (int r = 0; r < 2; ++r) {
    (void)hipMemcpyAsync(yin, ydec, (size_t)NROW_ * DIM_ * 4, hipMemcpyDeviceToDevice, stream);
    for (int bl = 0; bl < 2; ++bl) {
      const int layer = r * 2 + bl;
      k_im2col<<<blocks((size_t)NROW_ * 3 * DIM_), TPB, 0, stream>>>(ydec, col);
      gemm(col, W2h + (size_t)layer * 1536 * 512, ytmp, nullptr,
           conv_b + layer * 512, NROW_, DIM_, 3 * DIM_, 0);
      k_gn_stats<<<B_ * 8, TPB, 0, stream>>>(ytmp, stats);
      k_gn_silu<<<blocks((size_t)NROW_ * DIM_), TPB, 0, stream>>>(
          ytmp, ydec, stats, gn_w + layer * 512, gn_b + layer * 512);
    }
    k_add<<<blocks((size_t)NROW_ * DIM_), TPB, 0, stream>>>(ydec, yin, (size_t)NROW_ * DIM_);
  }

  // ---- logits + smoothed cross entropy ----
  k_cast16<<<blocks((size_t)NROW_ * DIM_), TPB, 0, stream>>>(ydec, dech, (size_t)NROW_ * DIM_);
  gemm(dech, Wlogh, logits, nullptr, b_logits, NROW_, 9 * ND_, DIM_, 0);
  k_ce<<<(NROW_ * 9) / 8, TPB, 0, stream>>>(logits, fco, scal + 2);
  k_final<<<1, 1, 0, stream>>>(scal, (float*)d_out);
}
